// Attention_58248346469030
// MI455X (gfx1250) — compile-verified, matrix operand order checked by reading, not run on hardware
//
#include <hip/hip_runtime.h>
#include <hip/hip_bf16.h>
#include <math.h>

// ---------------------------------------------------------------------------
// Attention (no head split, per reference): out = softmax(xWq (xWk)^T * s) xWv Wout + b
// Pipeline: [qkv GEMM] -> [Q K^T GEMM, fp32 scores] -> [row softmax -> bf16 P
// in place] -> [P V GEMM] -> [out-proj GEMM + bias].
// All GEMMs: bf16 WMMA (v_wmma_f32_16x16x32_bf16), fp32 accumulate,
// double-buffered LDS with batched global loads (load-all / wait-once / store).
// ---------------------------------------------------------------------------

typedef __attribute__((ext_vector_type(16))) __bf16 v16bf;
typedef __attribute__((ext_vector_type(8)))  float  v8f;

#define BM 128
#define BN 64
#define BK 64
#define LDA_P 80   // padded LDS row stride in bf16 elems (160B, 32B-aligned frags)
#define LDB_P 80
#define NTHREADS 256

__device__ __forceinline__ unsigned short f2bf(float f) {
  unsigned u = __builtin_bit_cast(unsigned, f);
  u += 0x7FFFu + ((u >> 16) & 1u);          // round-to-nearest-even
  return (unsigned short)(u >> 16);
}

// A: MxK row-major (f32 or bf16).  B: KxN row-major (f32 or bf16) unless B_NK,
// in which case B is NxK row-major bf16 (used for Q @ K^T: B rows are keys).
// C: f32 or bf16, optional scale/bias.
template <bool A_BF16, bool B_BF16, bool B_NK, bool OUT_BF16, bool BIAS>
__global__ __launch_bounds__(NTHREADS)
void gemm_wmma_bf16(const void* __restrict__ Aptr, const void* __restrict__ Bptr,
                    void* __restrict__ Cptr, const float* __restrict__ bias,
                    float scale, int K, int lda, int ldb, int ldc,
                    long aBatch, long bBatch, long cBatch) {
  __shared__ __align__(32) unsigned short sa[2][BM * LDA_P];
  __shared__ __align__(32) unsigned short sb[2][BN * LDB_P];

  const int tid  = threadIdx.x;
  const int lane = tid & 31;
  const int wid  = tid >> 5;     // 8 waves
  const int wm   = wid >> 1;     // 0..3 : 32-row slab
  const int wn   = wid & 1;      // 0..1 : 32-col slab
  const int l15  = lane & 15;
  const int lhi  = lane >> 4;

  const int m0 = blockIdx.y * BM;
  const int n0 = blockIdx.x * BN;
  const int b  = blockIdx.z;

  const v8f vzero = {0.f, 0.f, 0.f, 0.f, 0.f, 0.f, 0.f, 0.f};
  v8f acc[2][2];
  acc[0][0] = vzero; acc[0][1] = vzero; acc[1][0] = vzero; acc[1][1] = vzero;

  // Register staging buffers (unused flavors are dead-code eliminated).
  float4 aF[8]; uint4 aU[4];
  float4 bF[4]; uint4 bU[2];

  // ---- phase 1: issue ALL global loads for one K-tile (no uses -> batched) --
  auto globalLoad = [&](int kt) {
    if (A_BF16) {
      const unsigned short* Ag = (const unsigned short*)Aptr + (size_t)b * aBatch;
#pragma unroll
      for (int p = 0; p < 4; ++p) {
        int linear = (p * NTHREADS + tid) * 8;
        int row = linear / BK, col = linear % BK;
        aU[p] = *(const uint4*)(Ag + (size_t)(m0 + row) * lda + kt + col);
      }
    } else {
      const float* Ag = (const float*)Aptr + (size_t)b * aBatch;
#pragma unroll
      for (int p = 0; p < 8; ++p) {
        int linear = (p * NTHREADS + tid) * 4;
        int row = linear / BK, col = linear % BK;
        aF[p] = *(const float4*)(Ag + (size_t)(m0 + row) * lda + kt + col);
      }
    }
    if (B_NK) {  // NxK bf16 (key matrix)
      const unsigned short* Bg = (const unsigned short*)Bptr + (size_t)b * bBatch;
#pragma unroll
      for (int p = 0; p < 2; ++p) {
        int linear = (p * NTHREADS + tid) * 8;
        int n = linear / BK, col = linear % BK;
        bU[p] = *(const uint4*)(Bg + (size_t)(n0 + n) * ldb + kt + col);
      }
    } else if (B_BF16) {  // KxN bf16 (V)
      const unsigned short* Bg = (const unsigned short*)Bptr + (size_t)b * bBatch;
#pragma unroll
      for (int p = 0; p < 2; ++p) {
        int linear = (p * NTHREADS + tid) * 8;
        int krow = linear / BN, nc = linear % BN;
        bU[p] = *(const uint4*)(Bg + (size_t)(kt + krow) * ldb + n0 + nc);
      }
    } else {  // KxN f32 (weights)
      const float* Bg = (const float*)Bptr + (size_t)b * bBatch;
#pragma unroll
      for (int p = 0; p < 4; ++p) {
        int linear = (p * NTHREADS + tid) * 4;
        int krow = linear / BN, nc = linear % BN;
        bF[p] = *(const float4*)(Bg + (size_t)(kt + krow) * ldb + n0 + nc);
      }
    }
  };

  // ---- phase 2: convert (if f32) and store the staged tile into LDS --------
  // A tile: [row][k] bf16.  B tile: [n][k] bf16 (transposed here if K-major).
  auto ldsStore = [&](unsigned short* la, unsigned short* lb) {
    if (A_BF16) {
#pragma unroll
      for (int p = 0; p < 4; ++p) {
        int linear = (p * NTHREADS + tid) * 8;
        int row = linear / BK, col = linear % BK;
        *(uint4*)(&la[row * LDA_P + col]) = aU[p];
      }
    } else {
#pragma unroll
      for (int p = 0; p < 8; ++p) {
        int linear = (p * NTHREADS + tid) * 4;
        int row = linear / BK, col = linear % BK;
        unsigned short* d = &la[row * LDA_P + col];
        d[0] = f2bf(aF[p].x); d[1] = f2bf(aF[p].y);
        d[2] = f2bf(aF[p].z); d[3] = f2bf(aF[p].w);
      }
    }
    if (B_NK) {
#pragma unroll
      for (int p = 0; p < 2; ++p) {
        int linear = (p * NTHREADS + tid) * 8;
        int n = linear / BK, col = linear % BK;
        *(uint4*)(&lb[n * LDB_P + col]) = bU[p];
      }
    } else if (B_BF16) {
#pragma unroll
      for (int p = 0; p < 2; ++p) {
        int linear = (p * NTHREADS + tid) * 8;
        int krow = linear / BN, nc = linear % BN;
        const unsigned short* e = (const unsigned short*)&bU[p];
#pragma unroll
        for (int i = 0; i < 8; ++i) lb[(nc + i) * LDB_P + krow] = e[i];
      }
    } else {
#pragma unroll
      for (int p = 0; p < 4; ++p) {
        int linear = (p * NTHREADS + tid) * 4;
        int krow = linear / BN, nc = linear % BN;
        lb[(nc + 0) * LDB_P + krow] = f2bf(bF[p].x);
        lb[(nc + 1) * LDB_P + krow] = f2bf(bF[p].y);
        lb[(nc + 2) * LDB_P + krow] = f2bf(bF[p].z);
        lb[(nc + 3) * LDB_P + krow] = f2bf(bF[p].w);
      }
    }
  };

  // ---- WMMA over one staged K-tile -----------------------------------------
  // A frag: lane l15 = row, lhi picks K half (16 contiguous bf16 = 32B).
  // B frag: lane l15 = col n, lhi picks K half (LDS tile is [n][k]).
  auto compute = [&](const unsigned short* la, const unsigned short* lb) {
#pragma unroll
    for (int kk = 0; kk < BK; kk += 32) {
      v16bf afr[2], bfr[2];
#pragma unroll
      for (int r = 0; r < 2; ++r)
        afr[r] = *(const v16bf*)(&la[(wm * 32 + r * 16 + l15) * LDA_P + kk + lhi * 16]);
#pragma unroll
      for (int c = 0; c < 2; ++c)
        bfr[c] = *(const v16bf*)(&lb[(wn * 32 + c * 16 + l15) * LDB_P + kk + lhi * 16]);
#pragma unroll
      for (int r = 0; r < 2; ++r)
#pragma unroll
        for (int c = 0; c < 2; ++c)
          acc[r][c] = __builtin_amdgcn_wmma_f32_16x16x32_bf16(
              false, afr[r], false, bfr[c], (short)0, acc[r][c], false, false);
    }
  };

  // ---- main loop: double-buffered, one barrier per K-step ------------------
  globalLoad(0);
  ldsStore(sa[0], sb[0]);
  __syncthreads();

  for (int kt = 0; kt < K; kt += BK) {
    const int cur = (kt / BK) & 1;
    const bool hasNext = (kt + BK) < K;
    if (hasNext) globalLoad(kt + BK);        // overlap HBM with WMMA below
    compute(sa[cur], sb[cur]);
    if (hasNext) ldsStore(sa[cur ^ 1], sb[cur ^ 1]);
    __syncthreads();
  }

  // ------------------------------- epilogue ---------------------------------
  // C/D layout: VGPR i, lanes 0-15 -> M = i, lanes 16-31 -> M = 8 + i.
  const int gm = m0 + wm * 32;
  const int gn = n0 + wn * 32;
#pragma unroll
  for (int r = 0; r < 2; ++r) {
#pragma unroll
    for (int c = 0; c < 2; ++c) {
      const int colg = gn + c * 16 + l15;
      const float bv = BIAS ? bias[colg] : 0.f;
#pragma unroll
      for (int i = 0; i < 8; ++i) {
        const int rowg = gm + r * 16 + lhi * 8 + i;
        const float val = acc[r][c][i] * scale + bv;
        const size_t idx = (size_t)b * cBatch + (size_t)rowg * ldc + colg;
        if (OUT_BF16) ((unsigned short*)Cptr)[idx] = f2bf(val);
        else          ((float*)Cptr)[idx] = val;
      }
    }
  }
}

// Row softmax over 2048 fp32 scores, rewritten IN PLACE as bf16 P.
// Whole row is read into registers (8/thread) before any write, so aliasing is
// safe; the shared-memory reductions order the global reads before the writes.
__global__ __launch_bounds__(256)
void softmax_row_inplace_bf16(float* __restrict__ S, int n) {
  const int row = blockIdx.x;
  float* src = S + (size_t)row * n;
  unsigned short* dst = (unsigned short*)src;
  const int tid = threadIdx.x;

  float v[8];
  float m = -1e30f;
#pragma unroll
  for (int i = 0; i < 8; ++i) { v[i] = src[tid + i * 256]; m = fmaxf(m, v[i]); }

  __shared__ float red[256];
  red[tid] = m;
  __syncthreads();
  for (int s = 128; s > 0; s >>= 1) {
    if (tid < s) red[tid] = fmaxf(red[tid], red[tid + s]);
    __syncthreads();
  }
  m = red[0];
  __syncthreads();

  float sum = 0.f;
#pragma unroll
  for (int i = 0; i < 8; ++i) { v[i] = __expf(v[i] - m); sum += v[i]; }
  red[tid] = sum;
  __syncthreads();
  for (int s = 128; s > 0; s >>= 1) {
    if (tid < s) red[tid] += red[tid + s];
    __syncthreads();
  }
  const float inv = 1.0f / red[0];

#pragma unroll
  for (int i = 0; i < 8; ++i) dst[tid + i * 256] = f2bf(v[i] * inv);
}

// ---------------------------------------------------------------------------
extern "C" void kernel_launch(void* const* d_in, const int* in_sizes, int n_in,
                              void* d_out, int out_size, void* d_ws, size_t ws_size,
                              hipStream_t stream) {
  const float* x    = (const float*)d_in[0];  // [4,2048,1024]
  const float* Wqkv = (const float*)d_in[1];  // [1024,3072]
  const float* Wout = (const float*)d_in[2];  // [1024,1024]
  const float* bout = (const float*)d_in[3];  // [1024]
  float* out = (float*)d_out;                 // [4,2048,1024] f32

  const int B = 4, S = 2048, D = 1024, N3 = 3072, MS = B * S;  // MS = 8192

  // Workspace layout (~134 MB):
  //   qkv bf16 [8192 x 3072], scores f32 [4 x 2048 x 2048] (reused in place as
  //   bf16 P with ushort row stride 4096), attn-out bf16 [8192 x 1024].
  char* ws = (char*)d_ws;
  unsigned short* qkv = (unsigned short*)ws;
  float* scores = (float*)(ws + (size_t)MS * N3 * 2);
  unsigned short* ao =
      (unsigned short*)((char*)scores + (size_t)B * S * S * 4);

  dim3 blk(NTHREADS);

  // 1) qkv = x @ Wqkv            (A f32, B KxN f32, out bf16)
  gemm_wmma_bf16<false, false, false, true, false>
      <<<dim3(N3 / BN, MS / BM, 1), blk, 0, stream>>>(
          x, Wqkv, qkv, nullptr, 1.0f, D, D, N3, N3, 0, 0, 0);

  // 2) scores[b] = q[b] @ k[b]^T * 0.125   (A bf16, B NxK bf16, out f32)
  gemm_wmma_bf16<true, true, true, false, false>
      <<<dim3(S / BN, S / BM, B), blk, 0, stream>>>(
          qkv, qkv + D, scores, nullptr, 0.125f, D, N3, N3, S,
          (long)S * N3, (long)S * N3, (long)S * S);

  // 3) row softmax, in place fp32 -> bf16 P
  softmax_row_inplace_bf16<<<dim3(MS), blk, 0, stream>>>(scores, S);

  // 4) ao[b] = P[b] @ v[b]       (A bf16 stride 4096, B KxN bf16, out bf16)
  gemm_wmma_bf16<true, true, false, true, false>
      <<<dim3(D / BN, S / BM, B), blk, 0, stream>>>(
          (unsigned short*)scores, qkv + 2 * D, ao, nullptr, 1.0f, S,
          2 * S, N3, D, (long)S * 2 * S, (long)S * N3, (long)S * D);

  // 5) out = ao @ Wout + b_out   (A bf16, B KxN f32, out f32 + bias)
  gemm_wmma_bf16<true, false, false, false, true>
      <<<dim3(D / BN, MS / BM, 1), blk, 0, stream>>>(
          ao, Wout, out, bout, 1.0f, D, D, D, D, 0, 0, 0);
}
